// MultiScaleDeformableAttention_87978110091279
// MI455X (gfx1250) — matrix operand order repaired and checked by reading, not compiled, not run on hardware
//
#include <hip/hip_runtime.h>
#include <hip/hip_bf16.h>

// MS-Deformable-Attention forward for MI455X (gfx1250, wave32, WMMA + TDM).
//
// Pipeline (all scratch in d_ws):
//   0) convert the 4 weight matrices to f16 (<=128KB each, reused by every block)
//   1) value_proj = value_levels @ W_val^T + b_val        (WMMA GEMM + TDM B-tiles)
//   2) off_raw    = query @ W_off^T + b_off               (WMMA GEMM + TDM B-tiles)
//   3) attn_raw   = query @ W_attn^T + b_attn             (WMMA GEMM + TDM B-tiles)
//   4) fused softmax + bilinear sampling -> out_heads     (gather kernel, L2-resident value)
//   5) d_out      = out_heads @ W_out^T + b_out           (WMMA GEMM + TDM B-tiles)

typedef __attribute__((ext_vector_type(16))) _Float16     v16h;
typedef __attribute__((ext_vector_type(8)))  float        v8f;
typedef __attribute__((ext_vector_type(4)))  unsigned int u32x4;
typedef __attribute__((ext_vector_type(8)))  int          i32x8;
typedef __attribute__((ext_vector_type(4)))  int          i32x4;

#define NHEADS 8
#define NLVL   4
#define NPTS   4
#define DH     32
#define CDIM   256
#define S_TOT  19947   // 100*150 + 50*75 + 25*38 + 13*19

// ---------------------------------------------------------------------------
// Tensor Data Mover wrapper (arity differs between ROCm7.2/clang-22 and
// amdgpu-toolchain/clang-23; see CDNA5_HIP.md).
// ---------------------------------------------------------------------------
__device__ __forceinline__ void tdm_load_to_lds(u32x4 g0, i32x8 g1) {
  i32x4 z4 = {0, 0, 0, 0};
#if defined(__clang_major__) && (__clang_major__ >= 23)
  i32x8 z8 = {0, 0, 0, 0, 0, 0, 0, 0};
  __builtin_amdgcn_tensor_load_to_lds(g0, g1, z4, z4, z8, 0);
#else
  __builtin_amdgcn_tensor_load_to_lds(g0, g1, z4, z4, 0);
#endif
}

// ---------------------------------------------------------------------------
// Weight conversion: f32 [n] -> f16 [n]   (n % 4 == 0)
// ---------------------------------------------------------------------------
__global__ __launch_bounds__(256) void cvt_w_f16(const float* __restrict__ in,
                                                 _Float16* __restrict__ out, int n) {
  const int i = (blockIdx.x * 256 + threadIdx.x) * 4;
  if (i + 3 < n) {
    const float4 v = *(const float4*)(in + i);
    out[i + 0] = (_Float16)v.x; out[i + 1] = (_Float16)v.y;
    out[i + 2] = (_Float16)v.z; out[i + 3] = (_Float16)v.w;
  }
}

// ---------------------------------------------------------------------------
// GEMM: out[M,N] = A[M,K] @ Wh[N,K]^T + bias[N]
//   A fp32 (converted to f16 while staging to LDS), Wh f16 (TDM -> LDS).
// Block: 256 threads = 8 waves; tile 64(M) x 128(N); K-step 64, double buffered.
// Wave (wm 0..3, wn 0..1) computes a 16x64 strip = 4 WMMA tiles; 8 WMMA/K-step.
// Requires: K % 64 == 0, N % 128 == 0  (K=256, N in {128,256} here).
// ---------------------------------------------------------------------------
__global__ __launch_bounds__(256) void gemm_bias_wmma(
    const float* __restrict__ A, const _Float16* __restrict__ Wh,
    const float* __restrict__ bias, float* __restrict__ out,
    int M, int N, int K)
{
  __shared__ _Float16 sA[2][64][64];    // 16 KB : activations (f32->f16 staged)
  __shared__ _Float16 sB[2][128][64];   // 32 KB : weights (TDM direct)

  const int tid  = threadIdx.x;
  const int lane = tid & 31;
  const int wave = tid >> 5;
  const int wm   = wave >> 1;            // 0..3 -> 16-row slab
  const int wn   = wave & 1;             // 0..1 -> 64-col slab
  const int row0 = blockIdx.x * 64;
  const int colB = blockIdx.y * 128;
  const int m    = lane & 15;
  const int hi   = lane >> 4;
  const int nk   = K >> 6;

  v8f acc[4] = {v8f{}, v8f{}, v8f{}, v8f{}};

  // --- TDM: async-load one 128(N) x 64(K) f16 weight tile into sB[buf] ---
  auto issue_tdm = [&](int kt, int buf) {
    const uint64_t ga =
        (uint64_t)(uintptr_t)(Wh + (size_t)colB * K + (size_t)kt * 64);
    u32x4 g0;
    g0.x = 1u;                                          // count=1, user mode
    g0.y = (unsigned)(uintptr_t)(&sB[buf][0][0]);       // lds_addr (bytes)
    g0.z = (unsigned)ga;                                // global_addr[31:0]
    g0.w = (unsigned)((ga >> 32) & 0x1FFFFFFu) | (2u << 30);  // addr[56:32]|type=2
    i32x8 g1;
    g1[0] = (1 << 16);                                  // data_size = 2 bytes
    g1[1] = (int)((unsigned)(K & 0xFFFF) << 16);        // tensor_dim0[15:0]
    g1[2] = (int)(((unsigned)K >> 16) |
                  ((unsigned)(N & 0xFFFF) << 16));      // dim0[31:16] | dim1[15:0]
    g1[3] = (64 << 16);                                 // dim1[31:16]=0 | tile_dim0=64
    g1[4] = 128;                                        // tile_dim1=128 | tile_dim2=0
    g1[5] = K;                                          // tensor_dim0_stride[31:0]
    g1[6] = 0;
    g1[7] = 0;
    tdm_load_to_lds(g0, g1);
  };

  // --- stage one 64x64 fp32 A tile -> f16 LDS (16 elems / thread) ---
  auto stage_a = [&](int kt, int buf) {
    const int r  = tid >> 2;
    const int c0 = (tid & 3) * 16;
    const int gr = row0 + r;
    float4 v0 = make_float4(0.f, 0.f, 0.f, 0.f), v1 = v0, v2 = v0, v3 = v0;
    if (gr < M) {
      const float* p = A + (size_t)gr * K + kt * 64 + c0;
      v0 = ((const float4*)p)[0]; v1 = ((const float4*)p)[1];
      v2 = ((const float4*)p)[2]; v3 = ((const float4*)p)[3];
    }
    _Float16* d = &sA[buf][r][c0];
    d[0]  = (_Float16)v0.x; d[1]  = (_Float16)v0.y;
    d[2]  = (_Float16)v0.z; d[3]  = (_Float16)v0.w;
    d[4]  = (_Float16)v1.x; d[5]  = (_Float16)v1.y;
    d[6]  = (_Float16)v1.z; d[7]  = (_Float16)v1.w;
    d[8]  = (_Float16)v2.x; d[9]  = (_Float16)v2.y;
    d[10] = (_Float16)v2.z; d[11] = (_Float16)v2.w;
    d[12] = (_Float16)v3.x; d[13] = (_Float16)v3.y;
    d[14] = (_Float16)v3.z; d[15] = (_Float16)v3.w;
  };

  // prologue: kick off tile 0
  if (tid < 32) issue_tdm(0, 0);
  stage_a(0, 0);

  for (int kt = 0; kt < nk; ++kt) {
    const int cur = kt & 1;
    if (kt + 1 < nk) {
      if (tid < 32) issue_tdm(kt + 1, cur ^ 1);   // overlap next B tile (TDM)
      stage_a(kt + 1, cur ^ 1);                   // overlap next A tile
    }
    if (tid < 32) {                               // wave0: oldest TDM tile done
      if (kt + 1 < nk) __builtin_amdgcn_s_wait_tensorcnt(1);
      else             __builtin_amdgcn_s_wait_tensorcnt(0);
    }
    __syncthreads();                              // sA/sB[cur] visible to all

    // ---- 8 WMMAs on the resident buffers (CDNA5 fragment layouts) ----
#pragma unroll
    for (int kc = 0; kc < 2; ++kc) {
      v16h af;
#pragma unroll
      for (int e = 0; e < 8; ++e)  af[e] = sA[cur][wm * 16 + m][kc * 32 + hi * 8 + e];
#pragma unroll
      for (int e = 8; e < 16; ++e) af[e] = sA[cur][wm * 16 + m][kc * 32 + hi * 8 + 8 + e];
#pragma unroll
      for (int t = 0; t < 4; ++t) {
        v16h bf;
#pragma unroll
        for (int e = 0; e < 16; ++e)
          bf[e] = sB[cur][wn * 64 + t * 16 + m][kc * 32 + hi * 8 + e + (e >= 8 ? 8 : 0)];
        acc[t] = __builtin_amdgcn_wmma_f32_16x16x32_f16(
            false, af, false, bf, (short)0, acc[t], false, false);
      }
    }
    __syncthreads();                              // protect buffers for re-fill
  }

  // ---- store: VGPR r holds (M = r + 8*(lane>=16), N = lane&15) ----
#pragma unroll
  for (int t = 0; t < 4; ++t) {
    const int n  = colB + wn * 64 + t * 16 + m;
    const float bv = bias[n];
#pragma unroll
    for (int r = 0; r < 8; ++r) {
      const int mrow = row0 + wm * 16 + hi * 8 + r;
      if (mrow < M) out[(size_t)mrow * N + n] = acc[t][r] + bv;
    }
  }
}

// ---------------------------------------------------------------------------
// Fused softmax + bilinear sampling.
// Block (32, 8): lanes = dh channels, rows = heads; one block per (b, q).
// value: [B, S, NH, DH], off_raw: [B*Nq, 256], attn_raw: [B*Nq, 128],
// ref: [B*Nq, NL, 2], out_heads: [B*Nq, 256].
// ---------------------------------------------------------------------------
__global__ __launch_bounds__(256) void msda_sample(
    const float* __restrict__ value, const float* __restrict__ off_raw,
    const float* __restrict__ attn_raw, const float* __restrict__ ref,
    float* __restrict__ out_heads, int Nq)
{
  const int lane = threadIdx.x;        // 0..31
  const int h    = threadIdx.y;        // 0..7
  const int bq   = blockIdx.x;
  const int b    = bq / Nq;

  const float* offp = off_raw  + (size_t)bq * 256 + h * 32;  // [NL,NP,2]
  const float* ap   = attn_raw + (size_t)bq * 128 + h * 16;  // [NL*NP]

  // ---- softmax over 16 logits, duplicated in both 16-lane halves ----
  const int j = lane & 15;
  float logit = ap[j];
  float mx = logit;
#pragma unroll
  for (int d = 1; d < 16; d <<= 1) mx = fmaxf(mx, __shfl_xor(mx, d));
  float ex = __expf(logit - mx);
  float sm = ex;
#pragma unroll
  for (int d = 1; d < 16; d <<= 1) sm += __shfl_xor(sm, d);
  const float wsm = ex / sm;

  // ---- per-sample pixel coords held by lane j (duplicated halves) ----
  const int jl  = j >> 2;
  const int Wl0 = (jl == 0) ? 150 : (jl == 1) ? 75 : (jl == 2) ? 38 : 19;
  const int Hl0 = (jl == 0) ? 100 : (jl == 1) ? 50 : (jl == 2) ? 25 : 13;
  const float rx = ref[((size_t)bq * NLVL + jl) * 2 + 0];
  const float ry = ref[((size_t)bq * NLVL + jl) * 2 + 1];
  const float px = rx * (float)Wl0 + offp[2 * j + 0] - 0.5f;  // (rx + ox/W)*W - 0.5
  const float py = ry * (float)Hl0 + offp[2 * j + 1] - 0.5f;

  float acc = 0.f;
#pragma unroll
  for (int jj = 0; jj < 16; ++jj) {
    const int l  = jj >> 2;
    const int Wl = (l == 0) ? 150 : (l == 1) ? 75 : (l == 2) ? 38 : 19;
    const int Hl = (l == 0) ? 100 : (l == 1) ? 50 : (l == 2) ? 25 : 13;
    const int st = (l == 0) ? 0 : (l == 1) ? 15000 : (l == 2) ? 18750 : 19700;

    const float pxj = __shfl(px, jj);
    const float pyj = __shfl(py, jj);
    const float wj  = __shfl(wsm, jj);

    const float x0f = floorf(pxj), y0f = floorf(pyj);
    const int ix = (int)x0f, iy = (int)y0f;
    const float fx = pxj - x0f, fy = pyj - y0f;

    const float* base =
        value + (((size_t)b * S_TOT + st) * NHEADS + h) * DH + lane;

    // wave-uniform bounds (ix/iy identical across lanes)
    const bool vx0 = (ix >= 0) && (ix < Wl);
    const bool vx1 = (ix + 1 >= 0) && (ix + 1 < Wl);
    const bool vy0 = (iy >= 0) && (iy < Hl);
    const bool vy1 = (iy + 1 >= 0) && (iy + 1 < Hl);

    float s00 = 0.f, s10 = 0.f, s01 = 0.f, s11 = 0.f;
    if (vy0) {
      const float* rowp = base + (size_t)iy * Wl * (NHEADS * DH);
      if (vx0) s00 = rowp[(size_t)ix * (NHEADS * DH)];
      if (vx1) s10 = rowp[(size_t)(ix + 1) * (NHEADS * DH)];
    }
    if (vy1) {
      const float* rowp = base + (size_t)(iy + 1) * Wl * (NHEADS * DH);
      if (vx0) s01 = rowp[(size_t)ix * (NHEADS * DH)];
      if (vx1) s11 = rowp[(size_t)(ix + 1) * (NHEADS * DH)];
    }
    const float bil = (1.f - fx) * (1.f - fy) * s00 + fx * (1.f - fy) * s10 +
                      (1.f - fx) * fy * s01 + fx * fy * s11;
    acc = fmaf(wj, bil, acc);
  }

  out_heads[(size_t)bq * CDIM + h * DH + lane] = acc;
}

// ---------------------------------------------------------------------------
extern "C" void kernel_launch(void* const* d_in, const int* in_sizes, int n_in,
                              void* d_out, int out_size, void* d_ws, size_t ws_size,
                              hipStream_t stream) {
  const float* query        = (const float*)d_in[0];
  const float* value_levels = (const float*)d_in[1];
  const float* refpts       = (const float*)d_in[2];
  // d_in[3] spatial_shapes: compile-time constants here
  const float* W_off  = (const float*)d_in[4];
  const float* b_off  = (const float*)d_in[5];
  const float* W_attn = (const float*)d_in[6];
  const float* b_attn = (const float*)d_in[7];
  const float* W_val  = (const float*)d_in[8];
  const float* b_val  = (const float*)d_in[9];
  const float* W_out  = (const float*)d_in[10];
  const float* b_out  = (const float*)d_in[11];

  const int B  = 4, Nq = 8000;
  const int MQ = B * Nq;      // 32000
  const int MV = B * S_TOT;   // 79788

  // workspace: value_proj | off_raw | attn_raw | out_heads | f16 weights
  float* value_proj = (float*)d_ws;
  float* off_raw    = value_proj + (size_t)MV * CDIM;
  float* attn_raw   = off_raw    + (size_t)MQ * CDIM;
  float* out_heads  = attn_raw   + (size_t)MQ * 128;
  _Float16* whV = (_Float16*)(out_heads + (size_t)MQ * CDIM);
  _Float16* whO = whV + (size_t)CDIM * CDIM;
  _Float16* whA = whO + (size_t)CDIM * CDIM;
  _Float16* whU = whA + (size_t)128 * CDIM;

  const dim3 blk(256);

  // 0) weight conversion (tiny)
  cvt_w_f16<<<dim3(64), blk, 0, stream>>>(W_val,  whV, CDIM * CDIM);
  cvt_w_f16<<<dim3(64), blk, 0, stream>>>(W_off,  whO, CDIM * CDIM);
  cvt_w_f16<<<dim3(32), blk, 0, stream>>>(W_attn, whA, 128 * CDIM);
  cvt_w_f16<<<dim3(64), blk, 0, stream>>>(W_out,  whU, CDIM * CDIM);

  // 1) value projection
  gemm_bias_wmma<<<dim3((MV + 63) / 64, CDIM / 128), blk, 0, stream>>>(
      value_levels, whV, b_val, value_proj, MV, CDIM, CDIM);
  // 2) sampling offsets
  gemm_bias_wmma<<<dim3((MQ + 63) / 64, CDIM / 128), blk, 0, stream>>>(
      query, whO, b_off, off_raw, MQ, CDIM, CDIM);
  // 3) attention logits
  gemm_bias_wmma<<<dim3((MQ + 63) / 64, 1), blk, 0, stream>>>(
      query, whA, b_attn, attn_raw, MQ, 128, CDIM);
  // 4) fused softmax + bilinear sampling
  msda_sample<<<dim3(MQ), dim3(32, 8), 0, stream>>>(
      value_proj, off_raw, attn_raw, refpts, out_heads, Nq);
  // 5) output projection
  gemm_bias_wmma<<<dim3((MQ + 63) / 64, CDIM / 128), blk, 0, stream>>>(
      out_heads, whU, b_out, (float*)d_out, MQ, CDIM, CDIM);
}
